// ConvolutionalAttention_54503134986462
// MI455X (gfx1250) — matrix operand order repaired and checked by reading, not compiled, not run on hardware
//
#include <hip/hip_runtime.h>

#define B_  2
#define S_  2048
#define E_  1024
#define H_  16
#define DH_ 64

typedef __attribute__((ext_vector_type(16))) _Float16 v16h;
typedef __attribute__((ext_vector_type(8)))  _Float16 v8h;
typedef __attribute__((ext_vector_type(8)))  float    v8f;
typedef __attribute__((ext_vector_type(4)))  int      i32x4;
typedef __attribute__((ext_vector_type(8)))  int      i32x8;

__device__ __forceinline__ v8f wmma_f16(v16h a, v16h b, v8f c) {
  // D = A(16x32 f16) x B(32x16 f16) + C(16x16 f32)
  return __builtin_amdgcn_wmma_f32_16x16x32_f16(false, a, false, b, (short)0, c,
                                                false, false);
}

// 16 contiguous halfs (B-fragment: lane holds one column, 16 K values)
__device__ __forceinline__ v16h load16h(const _Float16* p) {
  v8h lo = *(const v8h*)p;
  v8h hi = *(const v8h*)(p + 8);
  v16h r;
#pragma unroll
  for (int i = 0; i < 8; ++i) { r[i] = lo[i]; r[i + 8] = hi[i]; }
  return r;
}

// A-fragment pattern: elements [0..7] <- p[0..7], [8..15] <- p[8..15]
__device__ __forceinline__ v16h load16h_afrag(const _Float16* p) {
  v8h lo = *(const v8h*)p;
  v8h hi = *(const v8h*)(p + 8);
  v16h r;
#pragma unroll
  for (int i = 0; i < 8; ++i) { r[i] = lo[i]; r[i + 8] = hi[i]; }
  return r;
}

// gfx1250 async DMA: copy 16 bytes global -> LDS, tracked by ASYNCcnt.
__device__ __forceinline__ void async_copy_b128(void* lds_ptr, const void* gptr) {
  unsigned lds_off = (unsigned)(unsigned long long)(uintptr_t)lds_ptr;
  asm volatile("global_load_async_to_lds_b128 %0, %1, off"
               :
               : "v"(lds_off), "v"((unsigned long long)(uintptr_t)gptr)
               : "memory");
}

// gfx1250 Tensor Data Mover: DMA a 2D f32 tile (rowLen-strided) into LDS with
// hardware row padding (32 DWORDs data + 4 DWORDs pad == LDW 36 layout).
// Tracked by TENSORcnt. EXEC-invariant; caller issues from one wave.
__device__ __forceinline__ void tdm_load_tile_f32(unsigned lds_off,
                                                  const void* gaddr,
                                                  int tileRows,
                                                  int rowLenElems) {
  unsigned long long ga = (unsigned long long)(uintptr_t)gaddr;
  i32x4 g0;
  g0[0] = 1;                                            // count=1 (valid D#)
  g0[1] = (int)lds_off;                                 // lds_addr
  g0[2] = (int)(unsigned)(ga & 0xFFFFFFFFull);          // global_addr[31:0]
  g0[3] = (int)(((ga >> 32) & 0x01FFFFFFull) | (2u << 30));  // addr[56:32]|type=2
  i32x8 g1;
  g1[0] = (2 << 16)    // data_size = 4B
        | (1 << 20)    // pad_enable
        | (4 << 22)    // pad_interval: 32 DWORDs
        | (3 << 25);   // pad_amount: 4 DWORDs
  g1[1] = (rowLenElems & 0xFFFF) << 16;                 // tensor_dim0[15:0]
  g1[2] = ((rowLenElems >> 16) & 0xFFFF)                // tensor_dim0[31:16]
        | ((tileRows & 0xFFFF) << 16);                  // tensor_dim1[15:0]
  g1[3] = ((tileRows >> 16) & 0xFFFF) | (32 << 16);     // tensor_dim1 hi | tile_dim0=32
  g1[4] = (tileRows & 0xFFFF);                          // tile_dim1 = rows
  g1[5] = rowLenElems;                                  // tensor_dim0_stride[31:0]
  g1[6] = 0;                                            // stride hi | dim1_stride lo
  g1[7] = 0;
  asm volatile("tensor_load_to_lds %0, %1" ::"s"(g0), "s"(g1) : "memory");
}

// ---------------------------------------------------------------------------
// WMMA GEMM:  out[m,n] = (sum_k A[m,k] * W[n,k] + bias[n]) * outScale
// MODE 0: out f16, head-major [B,H,S,DH]            (Q,K)
// MODE 1: out f16, head-major transposed [B,H,DH,S] (V^T)
// MODE 2: out f32, row-major [M,N]                  (final projection)
// f32 inputs: A tile via TDM tensor_load_to_lds (wave 0), W tile via
// GLOBAL_LOAD_ASYNC_TO_LDS_B128 (all threads); both double-buffered.
// ---------------------------------------------------------------------------
template <typename TA, int MODE>
__global__ __launch_bounds__(256) void gemm16(const TA* __restrict__ A,
                                              const float* __restrict__ W,
                                              const float* __restrict__ bias,
                                              void* __restrict__ out, int M,
                                              int Kd, int N, float outScale) {
  constexpr int LDW = 36;  // 144B rows: 16B-aligned b128 slots, conflict-free
  __shared__ float As[2][128][LDW];
  __shared__ float Bs[2][64][LDW];
  const int tid = threadIdx.x, w = tid >> 5, lane = tid & 31;
  const int m0 = blockIdx.x * 128, n0 = blockIdx.y * 64;
  const bool ASYNC = (sizeof(TA) == 4);

  v8f acc[4];
#pragma unroll
  for (int t = 0; t < 4; ++t)
#pragma unroll
    for (int i = 0; i < 8; ++i) acc[t][i] = 0.f;

  auto stage_A_tdm = [&](int buf, int kc) {  // wave 0 only
    tdm_load_tile_f32((unsigned)(unsigned long long)(uintptr_t)&As[buf][0][0],
                      A + (size_t)m0 * Kd + kc, 128, Kd);
  };
  auto stage_B_async = [&](int buf, int kc) {  // W: 64x32 f32 = 512 quads
#pragma unroll
    for (int i = 0; i < 2; ++i) {
      int q = i * 256 + tid, r = q >> 3, c = (q & 7) << 2;
      async_copy_b128(&Bs[buf][r][c], W + (size_t)(n0 + r) * Kd + kc + c);
    }
  };
  auto stage_manual = [&](int buf, int kc) {
#pragma unroll
    for (int i = 0; i < 16; ++i) {
      int idx = i * 256 + tid, r = idx >> 5, c = idx & 31;
      As[buf][r][c] = (float)A[(size_t)(m0 + r) * Kd + kc + c];
    }
#pragma unroll
    for (int i = 0; i < 8; ++i) {
      int idx = i * 256 + tid, r = idx >> 5, c = idx & 31;
      Bs[buf][r][c] = W[(size_t)(n0 + r) * Kd + kc + c];
    }
  };

  if (ASYNC) {
    if (w == 0) stage_A_tdm(0, 0);
    stage_B_async(0, 0);
  }

  for (int kc = 0; kc < Kd; kc += 32) {
    const int buf = (kc >> 5) & 1;
    if (ASYNC) {
      const bool more = (kc + 32 < Kd);
      if (more) {
        if (w == 0) stage_A_tdm(buf ^ 1, kc + 32);
        stage_B_async(buf ^ 1, kc + 32);
        asm volatile("s_wait_asynccnt 0x2" ::: "memory");
        if (w == 0) __builtin_amdgcn_s_wait_tensorcnt(1);
      } else {
        asm volatile("s_wait_asynccnt 0x0" ::: "memory");
        if (w == 0) __builtin_amdgcn_s_wait_tensorcnt(0);
      }
    } else {
      stage_manual(buf, kc);
    }
    __syncthreads();

    v16h a;
    {
      int row = (w << 4) + (lane & 15);
      int k0 = (lane >> 4) ? 16 : 0;
#pragma unroll
      for (int i = 0; i < 8; ++i) {
        a[i]     = (_Float16)As[buf][row][k0 + i];
        a[i + 8] = (_Float16)As[buf][row][k0 + 8 + i];
      }
    }
#pragma unroll
    for (int nt = 0; nt < 4; ++nt) {
      v16h bfr;
      int col = (nt << 4) + (lane & 15);
      int kk = (lane >> 4) << 4;
#pragma unroll
      for (int i = 0; i < 16; ++i) bfr[i] = (_Float16)Bs[buf][col][kk + i];
      acc[nt] = wmma_f16(a, bfr, acc[nt]);
    }
    __syncthreads();  // all reads of `buf` done -> next iter may restage it
  }

  const int rb = (lane >> 4) << 3;
#pragma unroll
  for (int nt = 0; nt < 4; ++nt) {
    int ncol = n0 + (nt << 4) + (lane & 15);
    float bv = bias[ncol];
#pragma unroll
    for (int r = 0; r < 8; ++r) {
      int m = m0 + (w << 4) + rb + r;
      float val = (acc[nt][r] + bv) * outScale;
      if (MODE == 0) {
        int bb = m / S_, s = m % S_, hh = ncol >> 6, d = ncol & 63;
        ((_Float16*)out)[(((size_t)bb * H_ + hh) * S_ + s) * DH_ + d] =
            (_Float16)val;
      } else if (MODE == 1) {
        int bb = m / S_, s = m % S_, hh = ncol >> 6, d = ncol & 63;
        ((_Float16*)out)[(((size_t)bb * H_ + hh) * DH_ + d) * S_ + s] =
            (_Float16)val;
      } else {
        ((float*)out)[(size_t)m * N + ncol] = val;
      }
    }
  }
}

// ---------------------------------------------------------------------------
// Fused conv-attention. Block = 512 threads = 16 waves, one wave per head.
// grid (S/16, B). Q is pre-scaled by 1/sqrt(DH) in its projection, so scores
// come out of the WMMA ready to store (no dependent VALU scaling chain).
// ---------------------------------------------------------------------------
__global__ __launch_bounds__(512) void conv_flash_attn(
    const _Float16* __restrict__ Qh, const _Float16* __restrict__ Kh,
    const _Float16* __restrict__ Vt, const float* __restrict__ conv_w,
    const float* __restrict__ conv_b, _Float16* __restrict__ attn_out) {
  constexpr int KT = 32;
  __shared__ float    raw[H_][16][KT + 2];  // col 0 = k0-1 halo, KT+1 = k0+KT halo
  __shared__ _Float16 ps[H_][16][KT];       // softmaxed probs (f16, A-frag layout)
  __shared__ float    m_run[H_][16];
  __shared__ float    l_run[H_][16];
  __shared__ float    scale_s[H_][16];
  __shared__ float    cb_s[H_];

  const int tid = threadIdx.x, w = tid >> 5, lane = tid & 31;
  const int h = w;                  // wave == head
  const int bb = blockIdx.y, q0 = blockIdx.x << 4;
  const int row = lane & 15;
  const int rb = (lane >> 4) << 3;  // C-layout row base for this half-wave
  const size_t headBase = ((size_t)bb * H_ + h) * (size_t)S_ * DH_;

  if (tid < H_ * 16) {
    m_run[tid >> 4][tid & 15] = -3.0e38f;
    l_run[tid >> 4][tid & 15] = 0.f;
  }
  if (tid < H_) cb_s[tid] = conv_b[tid];

  // conv weights for h_out == this wave's head (lane-uniform -> SGPRs)
  float cwr[H_][3];
#pragma unroll
  for (int hi = 0; hi < H_; ++hi)
#pragma unroll
    for (int t = 0; t < 3; ++t) cwr[hi][t] = conv_w[(h * H_ + hi) * 3 + t];

  // Q A-fragments for the 16-row query tile (dh chunks of 32)
  v16h qf[2];
#pragma unroll
  for (int f = 0; f < 2; ++f) {
    int d0 = (f << 5) + ((lane >> 4) ? 16 : 0);
    qf[f] = load16h_afrag(Qh + headBase + (size_t)(q0 + row) * DH_ + d0);
  }

  v8f acc[4];
#pragma unroll
  for (int t = 0; t < 4; ++t)
#pragma unroll
    for (int i = 0; i < 8; ++i) acc[t][i] = 0.f;

  __syncthreads();

  for (int k0 = 0; k0 < S_; k0 += KT) {
    // ---- batch-load all K B-fragments (one clause), then 4 WMMAs ----
    v16h kf[2][2];
#pragma unroll
    for (int nt = 0; nt < 2; ++nt) {
      int kcol = k0 + (nt << 4) + (lane & 15);
#pragma unroll
      for (int dc = 0; dc < 2; ++dc) {
        int d0 = (dc << 5) + ((lane >> 4) << 4);
        kf[nt][dc] = load16h(Kh + headBase + (size_t)kcol * DH_ + d0);
      }
    }
    // prefetch next K tile into WGP caches while we compute
    if (k0 + KT < S_)
      __builtin_prefetch(Kh + headBase + (size_t)(k0 + KT + row) * DH_, 0, 3);

    v8f sc[2];
#pragma unroll
    for (int nt = 0; nt < 2; ++nt) {
#pragma unroll
      for (int i = 0; i < 8; ++i) sc[nt][i] = 0.f;
#pragma unroll
      for (int dc = 0; dc < 2; ++dc) sc[nt] = wmma_f16(qf[dc], kf[nt][dc], sc[nt]);
    }
#pragma unroll
    for (int nt = 0; nt < 2; ++nt) {
      int kc = (nt << 4) + (lane & 15);
#pragma unroll
      for (int r = 0; r < 8; ++r) raw[h][rb + r][1 + kc] = sc[nt][r];
    }
    // ---- halo columns k0-1 / k0+KT via VALU dots + half-wave combine ----
    {
      float dL = 0.f, dR = 0.f;
      const bool hasL = (k0 > 0), hasR = (k0 + KT < S_);
#pragma unroll
      for (int f = 0; f < 2; ++f) {
        int d0 = (f << 5) + ((lane >> 4) ? 16 : 0);
        const _Float16* kl = Kh + headBase + (size_t)(k0 - 1) * DH_ + d0;
        const _Float16* kr = Kh + headBase + (size_t)(k0 + KT) * DH_ + d0;
#pragma unroll
        for (int i = 0; i < 8; ++i) {
          float q1 = (float)qf[f][i], q2 = (float)qf[f][i + 8];
          if (hasL) dL += q1 * (float)kl[i] + q2 * (float)kl[8 + i];
          if (hasR) dR += q1 * (float)kr[i] + q2 * (float)kr[8 + i];
        }
      }
      dL += __shfl_xor(dL, 16, 32);
      dR += __shfl_xor(dR, 16, 32);
      if (lane < 16) {
        raw[h][row][0]      = hasL ? dL : 0.f;
        raw[h][row][KT + 1] = hasR ? dR : 0.f;
      }
    }
    // ---- prefetch V B-fragments now; conv+softmax hides their latency ----
    v16h vf[4];
#pragma unroll
    for (int t = 0; t < 4; ++t) {
      int d = (t << 4) + (lane & 15);
      vf[t] = load16h(Vt + (((size_t)bb * H_ + h) * DH_ + d) * S_ + k0 +
                      ((lane >> 4) << 4));
    }
    __syncthreads();  // raw (all heads) ready for conv

    // ---- cross-head conv + online softmax; lane owns (q=row, 16 k's) ----
    {
      const int q = row;
      const int kb = (lane >> 4) << 4;
      float cacc[16];
#pragma unroll
      for (int i = 0; i < 16; ++i) cacc[i] = cb_s[h];
      for (int hi = 0; hi < H_; ++hi) {
        float rbuf[18];
#pragma unroll
        for (int j = 0; j < 18; ++j) rbuf[j] = raw[hi][q][kb + j];
        float w0 = cwr[hi][0], w1 = cwr[hi][1], w2 = cwr[hi][2];
#pragma unroll
        for (int i = 0; i < 16; ++i)
          cacc[i] += w0 * rbuf[i] + w1 * rbuf[i + 1] + w2 * rbuf[i + 2];
      }
      float mx = -3.0e38f;
#pragma unroll
      for (int i = 0; i < 16; ++i) mx = fmaxf(mx, cacc[i]);
      mx = fmaxf(mx, __shfl_xor(mx, 16, 32));  // full row max over the tile
      float m_old = m_run[h][q];
      float m_new = fmaxf(m_old, mx);
      float scl = __expf(m_old - m_new);
      float sum = 0.f;
#pragma unroll
      for (int i = 0; i < 16; ++i) {
        float p = __expf(cacc[i] - m_new);
        sum += p;
        ps[h][q][kb + i] = (_Float16)p;
      }
      sum += __shfl_xor(sum, 16, 32);
      if (lane < 16) {
        l_run[h][q] = l_run[h][q] * scl + sum;
        m_run[h][q] = m_new;
        scale_s[h][q] = scl;
      }
    }
    __syncthreads();  // ps/scale_s ready; raw free for next tile

    // ---- rescale accumulators (C-layout rows) + PV WMMAs ----
    {
      float scs[8];
#pragma unroll
      for (int r = 0; r < 8; ++r) scs[r] = scale_s[h][rb + r];
#pragma unroll
      for (int t = 0; t < 4; ++t)
#pragma unroll
        for (int r = 0; r < 8; ++r) acc[t][r] *= scs[r];

      int kk = (lane >> 4) ? 16 : 0;
      v16h pf = load16h_afrag(&ps[h][row][kk]);  // A: P(16x32)
#pragma unroll
      for (int t = 0; t < 4; ++t) acc[t] = wmma_f16(pf, vf[t], acc[t]);
    }
    __syncthreads();
  }

  // ---- finalize: divide by l, store merged-head f16 [B,S,E] ----
  {
    float linv[8];
#pragma unroll
    for (int r = 0; r < 8; ++r) linv[r] = 1.0f / l_run[h][rb + r];
#pragma unroll
    for (int t = 0; t < 4; ++t) {
      int d = (t << 4) + (lane & 15);
#pragma unroll
      for (int r = 0; r < 8; ++r) {
        int s = q0 + rb + r;
        attn_out[((size_t)bb * S_ + s) * E_ + h * DH_ + d] =
            (_Float16)(acc[t][r] * linv[r]);
      }
    }
  }
}

extern "C" void kernel_launch(void* const* d_in, const int* in_sizes, int n_in,
                              void* d_out, int out_size, void* d_ws,
                              size_t ws_size, hipStream_t stream) {
  (void)in_sizes; (void)n_in; (void)out_size; (void)ws_size;
  const float* query  = (const float*)d_in[0];
  const float* Wq     = (const float*)d_in[1];
  const float* bq     = (const float*)d_in[2];
  const float* Wk     = (const float*)d_in[3];
  const float* bk     = (const float*)d_in[4];
  const float* Wv     = (const float*)d_in[5];
  const float* bv     = (const float*)d_in[6];
  const float* Wo     = (const float*)d_in[7];
  const float* bo     = (const float*)d_in[8];
  const float* conv_w = (const float*)d_in[9];
  const float* conv_b = (const float*)d_in[10];

  const size_t headElems = (size_t)B_ * H_ * S_ * DH_;  // 4 Mi f16 each
  _Float16* qws  = (_Float16*)d_ws;
  _Float16* kws  = qws + headElems;
  _Float16* vtws = kws + headElems;
  _Float16* aws  = vtws + headElems;  // [B,S,E] f16 attention output

  dim3 gg((B_ * S_) / 128, E_ / 64);
  // Q projection pre-scaled by 1/sqrt(DH): scores need no post-WMMA scaling.
  gemm16<float, 0><<<gg, 256, 0, stream>>>(query, Wq, bq, qws, B_ * S_, E_, E_,
                                           0.125f);
  gemm16<float, 0><<<gg, 256, 0, stream>>>(query, Wk, bk, kws, B_ * S_, E_, E_,
                                           1.0f);
  gemm16<float, 1><<<gg, 256, 0, stream>>>(query, Wv, bv, vtws, B_ * S_, E_, E_,
                                           1.0f);
  conv_flash_attn<<<dim3(S_ / 16, B_), 512, 0, stream>>>(qws, kws, vtws, conv_w,
                                                         conv_b, aws);
  gemm16<_Float16, 2><<<gg, 256, 0, stream>>>(aws, Wo, bo, d_out, B_ * S_, E_,
                                              E_, 1.0f);
}
